// Prediction_48885317763373
// MI455X (gfx1250) — matrix operand order repaired and checked by reading, not compiled, not run on hardware
//
#include <hip/hip_runtime.h>
#include <math.h>

typedef __attribute__((ext_vector_type(16))) _Float16 v16h;
typedef __attribute__((ext_vector_type(4)))  _Float16 v4h;
typedef __attribute__((ext_vector_type(8)))  float    v8f;

#define BSZ 4
#define QL  128
#define CL  512
#define XD  64
#define ED  128
#define DM  128
#define YD  32

// ---------------- prep: qp = q @ W0[:XD] + b0  -> [B*QL, DM] ----------------
__global__ void qp_kernel(const float* __restrict__ q, const float* __restrict__ W0,
                          const float* __restrict__ b0, float* __restrict__ qp) {
    int idx = blockIdx.x * blockDim.x + threadIdx.x;     // < B*QL*DM
    int d  = idx & (DM - 1);
    int bq = idx >> 7;
    const float* qr = q + bq * XD;
    float s = b0[d];
#pragma unroll 8
    for (int x = 0; x < XD; ++x) s += qr[x] * W0[x * DM + d];
    qp[idx] = s;
}

// ---------------- prep: fp = f_embed @ W0[XD:] -> [B*CL, DM] ----------------
__global__ void fp_kernel(const float* __restrict__ f, const float* __restrict__ W0,
                          float* __restrict__ fp) {
    int idx = blockIdx.x * blockDim.x + threadIdx.x;     // < B*CL*DM
    int d  = idx & (DM - 1);
    int bc = idx >> 7;
    const float* fr = f + bc * ED;
    const float* w  = W0 + XD * DM;
    float s = 0.f;
#pragma unroll 8
    for (int e = 0; e < ED; ++e) s += fr[e] * w[e * DM + d];
    fp[idx] = s;
}

// ---------------- prep: row softmax of att_logits [B*QL, CL] ----------------
__global__ void softmax_kernel(const float* __restrict__ logits, float* __restrict__ att) {
    __shared__ float red[256];
    int row = blockIdx.x;
    int tid = threadIdx.x;
    const float* x = logits + (size_t)row * CL;
    float a = x[tid], b = x[tid + 256];
    red[tid] = fmaxf(a, b);
    __syncthreads();
    for (int off = 128; off > 0; off >>= 1) {
        if (tid < off) red[tid] = fmaxf(red[tid], red[tid + off]);
        __syncthreads();
    }
    float mx = red[0];
    __syncthreads();
    float e0 = __expf(a - mx), e1 = __expf(b - mx);
    red[tid] = e0 + e1;
    __syncthreads();
    for (int off = 128; off > 0; off >>= 1) {
        if (tid < off) red[tid] += red[tid + off];
        __syncthreads();
    }
    float inv = 1.f / red[0];
    att[(size_t)row * CL + tid]       = e0 * inv;
    att[(size_t)row * CL + tid + 256] = e1 * inv;
}

// ---- prep: repack W1 (128x128) and Wo (128x32) into f16 B-fragment order ----
// B 32x16 fragment, wave32: lane = g*16+n, element e -> K = s*32 + (e>>3)*16 + g*8 + (e&7), N = t*16 + n
__global__ void wconv_kernel(const float* __restrict__ W1, const float* __restrict__ Wo,
                             _Float16* __restrict__ W1h, _Float16* __restrict__ Woh) {
    int idx = blockIdx.x * blockDim.x + threadIdx.x;     // 0 .. 20479
    if (idx < 4 * 8 * 32 * 16) {                         // W1 fragments: s(4) x t(8)
        int frag = idx >> 9;         // s*8 + t
        int s = frag >> 3, t = frag & 7;
        int r = idx & 511;
        int lane = r >> 4, e = r & 15;
        int g = lane >> 4, n = lane & 15;
        int k = s * 32 + (e >> 3) * 16 + g * 8 + (e & 7);
        W1h[idx] = (_Float16)W1[k * DM + t * 16 + n];
    } else if (idx < 4 * 8 * 32 * 16 + 4 * 2 * 32 * 16) { // Wo fragments: s(4) x t2(2)
        int j = idx - 4 * 8 * 32 * 16;
        int frag = j >> 9;           // s*2 + t2
        int s = frag >> 1, t2 = frag & 1;
        int r = j & 511;
        int lane = r >> 4, e = r & 15;
        int g = lane >> 4, n = lane & 15;
        int k = s * 32 + (e >> 3) * 16 + g * 8 + (e & 7);
        Woh[j] = (_Float16)Wo[k * YD + t2 * 16 + n];
    }
}

// ---- A-fragment loader: 16-bit A 16x32 layout (ISA 7.12.2) from LDS tile [16][128] f16 ----
__device__ __forceinline__ v16h load_A_frag(const _Float16* tile, int s, int lane) {
    int m = lane & 15, g = lane >> 4;
    union { v16h v; unsigned u[8]; } r;
#pragma unroll
    for (int j = 0; j < 8; ++j) {
        int k = s * 32 + ((j >> 2) << 4) + (g << 3) + ((j & 3) << 1); // K,K+1 contiguous pair
        r.u[j] = *(const unsigned*)&tile[m * 128 + k];
    }
    return r.v;
}

// ------------------------------- fused main ---------------------------------
// grid = B*QL blocks, 256 threads (8 waves). Each wave: c-tiles of 16 rows.
__global__ void __launch_bounds__(256)
fused_mlp_attn_kernel(const float* __restrict__ qp, const float* __restrict__ fp,
                      const float* __restrict__ att,
                      const _Float16* __restrict__ W1h, const _Float16* __restrict__ Woh,
                      const float* __restrict__ b1, const float* __restrict__ bo,
                      float* __restrict__ out) {
    __shared__ __align__(16) _Float16 lds_tile[8 * 16 * 128]; // 4KB per wave
    __shared__ float y_acc[YD];

    const int bq   = blockIdx.x;
    const int b    = bq >> 7;           // bq / QL
    const int tid  = threadIdx.x;
    const int wave = tid >> 5;
    const int lane = tid & 31;
    const int g    = lane >> 4;
    const int n    = lane & 15;

    if (tid < YD) y_acc[tid] = 0.f;
    __syncthreads();

    const float* qp_row = qp + (size_t)bq * DM;
    const float* fp_b   = fp + (size_t)b * CL * DM;
    const float* att_row = att + (size_t)bq * CL;
    _Float16* tile = &lds_tile[wave * 16 * 128];
    const v16h* W1f = (const v16h*)W1h;   // fragment-ordered, 32B aligned
    const v16h* Wof = (const v16h*)Woh;

    // per-lane slice of qp row (b0 already folded in), reused by every c-tile
    float4 qk = *(const float4*)(qp_row + lane * 4);

    float yreg0 = 0.f, yreg1 = 0.f;

    for (int ct = wave; ct < CL / 16; ct += 8) {
        const int c0 = ct * 16;

        // ---- stage h0 = relu(qp + fp) as f16 [16][128] in LDS ----
#pragma unroll
        for (int row = 0; row < 16; ++row) {
            float4 fv = *(const float4*)(fp_b + (size_t)(c0 + row) * DM + lane * 4);
            float x0 = qk.x + fv.x, x1 = qk.y + fv.y, x2 = qk.z + fv.z, x3 = qk.w + fv.w;
            v4h hv;
            hv.x = (_Float16)(x0 > 0.f ? x0 : 0.f);
            hv.y = (_Float16)(x1 > 0.f ? x1 : 0.f);
            hv.z = (_Float16)(x2 > 0.f ? x2 : 0.f);
            hv.w = (_Float16)(x3 > 0.f ? x3 : 0.f);
            *(v4h*)&tile[row * 128 + lane * 4] = hv;
        }
        __builtin_amdgcn_wave_barrier();   // LDS is in-order within a wave; block reordering

        // ---- A fragments for h0 (K = 128 -> 4 steps of 32) ----
        v16h a0 = load_A_frag(tile, 0, lane);
        v16h a1 = load_A_frag(tile, 1, lane);
        v16h a2 = load_A_frag(tile, 2, lane);
        v16h a3 = load_A_frag(tile, 3, lane);

        // ---- GEMM1: h1 = relu(h0 @ W1 + b1), restage as f16 [16][128] ----
#pragma unroll
        for (int t = 0; t < 8; ++t) {
            v8f acc = {};
            acc = __builtin_amdgcn_wmma_f32_16x16x32_f16(false, a0, false, W1f[(0 * 8 + t) * 32 + lane], (short)0, acc, false, false);
            acc = __builtin_amdgcn_wmma_f32_16x16x32_f16(false, a1, false, W1f[(1 * 8 + t) * 32 + lane], (short)0, acc, false, false);
            acc = __builtin_amdgcn_wmma_f32_16x16x32_f16(false, a2, false, W1f[(2 * 8 + t) * 32 + lane], (short)0, acc, false, false);
            acc = __builtin_amdgcn_wmma_f32_16x16x32_f16(false, a3, false, W1f[(3 * 8 + t) * 32 + lane], (short)0, acc, false, false);
            float bias = b1[t * 16 + n];
#pragma unroll
            for (int j = 0; j < 8; ++j) {              // D layout: (M = j + 8g, N = n)
                float d = acc[j] + bias;
                d = d > 0.f ? d : 0.f;
                tile[(j + 8 * g) * 128 + t * 16 + n] = (_Float16)d;
            }
        }
        __builtin_amdgcn_wave_barrier();

        // ---- A fragments for h1 ----
        v16h h1a0 = load_A_frag(tile, 0, lane);
        v16h h1a1 = load_A_frag(tile, 1, lane);
        v16h h1a2 = load_A_frag(tile, 2, lane);
        v16h h1a3 = load_A_frag(tile, 3, lane);

        // ---- per-row attention weights for this lane's M half ----
        const float* ar = att_row + c0 + 8 * g;
        float4 wlo = *(const float4*)(ar);
        float4 whi = *(const float4*)(ar + 4);
        float w[8] = { wlo.x, wlo.y, wlo.z, wlo.w, whi.x, whi.y, whi.z, whi.w };

        // ---- GEMM2: yy = h1 @ Wo + bo, fold att weighting + M-reduction ----
#pragma unroll
        for (int t2 = 0; t2 < 2; ++t2) {
            v8f acc = {};
            acc = __builtin_amdgcn_wmma_f32_16x16x32_f16(false, h1a0, false, Wof[(0 * 2 + t2) * 32 + lane], (short)0, acc, false, false);
            acc = __builtin_amdgcn_wmma_f32_16x16x32_f16(false, h1a1, false, Wof[(1 * 2 + t2) * 32 + lane], (short)0, acc, false, false);
            acc = __builtin_amdgcn_wmma_f32_16x16x32_f16(false, h1a2, false, Wof[(2 * 2 + t2) * 32 + lane], (short)0, acc, false, false);
            acc = __builtin_amdgcn_wmma_f32_16x16x32_f16(false, h1a3, false, Wof[(3 * 2 + t2) * 32 + lane], (short)0, acc, false, false);
            float bias = bo[t2 * 16 + n];
            float s = 0.f;
#pragma unroll
            for (int j = 0; j < 8; ++j) s += w[j] * (acc[j] + bias);
            if (t2 == 0) yreg0 += s; else yreg1 += s;
        }
    }

    // combine the two M-halves (lane ^ 16 holds the other 8 rows of the same column n)
    yreg0 += __shfl_xor(yreg0, 16, 32);
    yreg1 += __shfl_xor(yreg1, 16, 32);
    if (g == 0) {
        atomicAdd(&y_acc[n], yreg0);
        atomicAdd(&y_acc[16 + n], yreg1);
    }
    __syncthreads();
    if (tid < YD) out[(size_t)bq * YD + tid] = y_acc[tid];
}

// ------------------------------- launcher -----------------------------------
extern "C" void kernel_launch(void* const* d_in, const int* in_sizes, int n_in,
                              void* d_out, int out_size, void* d_ws, size_t ws_size,
                              hipStream_t stream) {
    const float* q    = (const float*)d_in[0];   // [4,128,64]
    const float* femb = (const float*)d_in[1];   // [4,512,128]
    const float* attl = (const float*)d_in[2];   // [4,128,512]
    const float* W0   = (const float*)d_in[3];   // [192,128]
    const float* b0   = (const float*)d_in[4];   // [128]
    const float* W1   = (const float*)d_in[5];   // [128,128]
    const float* b1   = (const float*)d_in[6];   // [128]
    const float* Wo   = (const float*)d_in[7];   // [128,32]
    const float* bo   = (const float*)d_in[8];   // [32]
    float* out = (float*)d_out;                  // [4,128,32]

    // workspace layout (floats): qp | fp | att | W1h(f16) | Woh(f16)
    float* ws = (float*)d_ws;
    float* qp  = ws;                                   // 4*128*128   = 65536 f
    float* fp  = ws + 65536;                           // 4*512*128   = 262144 f
    float* att = ws + 65536 + 262144;                  // 4*128*512   = 262144 f
    _Float16* W1h = (_Float16*)(ws + 589824);          // 16384 halves
    _Float16* Woh = W1h + 4 * 8 * 32 * 16;             // 4096 halves

    qp_kernel<<<(BSZ * QL * DM) / 256, 256, 0, stream>>>(q, W0, b0, qp);
    fp_kernel<<<(BSZ * CL * DM) / 256, 256, 0, stream>>>(femb, W0, fp);
    softmax_kernel<<<BSZ * QL, 256, 0, stream>>>(attl, att);
    wconv_kernel<<<80, 256, 0, stream>>>(W1, Wo, W1h, Woh);
    fused_mlp_attn_kernel<<<BSZ * QL, 256, 0, stream>>>(qp, fp, att, W1h, Woh, b1, bo, out);
}
